// SpatialKDE_23519240913318
// MI455X (gfx1250) — matrix operand order repaired and checked by reading, not compiled, not run on hardware
//
#include <hip/hip_runtime.h>
#include <math.h>

// Problem geometry (fixed by the reference: b=2, g=8, feats 64x64/32x32/16x16)
#define N_ANCH 5376      // 4096 + 1024 + 256
#define NROW   16        // b*g rows
#define TOPK   10
#define NSLOT  160       // NROW * TOPK  (K dimension of the block-diagonal GEMM)
#define BW_F   0.15f
#define EPS_F  1e-9f

typedef __attribute__((ext_vector_type(16))) _Float16 v16h;
typedef __attribute__((ext_vector_type(8)))  _Float16 v8h;
typedef __attribute__((ext_vector_type(8)))  float    v8f;

// ws layout (floats): [0,160) slot weights | [160,320) slot anchor idx (int)
//                     [320,480) slot 1/(2h^2)
#define WS_W   0
#define WS_GI  NSLOT
#define WS_IV  (2 * NSLOT)

// ---------------------------------------------------------------------------
// Kernel 1: per-(b,g) top-10 of align_metric, bandwidth, per-slot tables,
// plus L2 warm-up prefetch of the selected dist-matrix rows (global_prefetch).
// ---------------------------------------------------------------------------
__global__ void kde_topk_kernel(const float* __restrict__ am,
                                const float* __restrict__ gtb,
                                const float* __restrict__ mgt,
                                const float* __restrict__ d0,
                                const float* __restrict__ d1,
                                const float* __restrict__ d2,
                                float* __restrict__ ws) {
  const int row = blockIdx.x;
  const int tid = threadIdx.x;
  const float* a = am + row * N_ANCH;

  __shared__ float s_val[256];
  __shared__ int   s_idx[256];
  __shared__ int   sel[TOPK];
  __shared__ float selv[TOPK];

  for (int it = 0; it < TOPK; ++it) {
    float bv = -__builtin_inff();
    int   bi = 0x7fffffff;
    for (int i = tid; i < N_ANCH; i += 256) {
      float x = a[i];
      bool taken = false;
      for (int j = 0; j < it; ++j) taken |= (sel[j] == i);
      if (!taken && (x > bv || (x == bv && i < bi))) { bv = x; bi = i; }
    }
    s_val[tid] = bv; s_idx[tid] = bi;
    __syncthreads();
    for (int s = 128; s > 0; s >>= 1) {
      if (tid < s) {
        float ov = s_val[tid + s]; int oi = s_idx[tid + s];
        if (ov > s_val[tid] || (ov == s_val[tid] && oi < s_idx[tid])) {
          s_val[tid] = ov; s_idx[tid] = oi;
        }
      }
      __syncthreads();
    }
    if (tid == 0) { sel[it] = s_idx[0]; selv[it] = s_val[0]; }
    __syncthreads();
  }

  const bool valid = mgt[row] > 0.0f;
  // bandwidth (computable by every thread; cheap)
  float w  = gtb[row * 4 + 2] - gtb[row * 4 + 0];
  float h  = gtb[row * 4 + 3] - gtb[row * 4 + 1];
  float hb = BW_F * sqrtf(w * h + EPS_F);
  float hsq = valid ? hb * hb : 1.0f;          // reference's invalid-row guard
  float inv = 1.0f / (2.0f * hsq);

  if (tid < TOPK) {
    int s = row * TOPK + tid;
    ws[WS_W + s] = valid ? selv[tid] : 0.0f;   // weight = am * topk-mask * valid
    ((int*)(ws + WS_GI))[s] = sel[tid];
    ws[WS_IV + s] = inv;
  }

  // Warm L2 with the 10 selected rows (<=160 KB/block): global_prefetch_b8.
  for (int j = 0; j < TOPK; ++j) {
    int gi = sel[j];
    const float* base; int nL, off;
    if (gi < 4096)      { base = d0; nL = 4096; off = 0; }
    else if (gi < 5120) { base = d1; nL = 1024; off = 4096; }
    else                { base = d2; nL = 256;  off = 5120; }
    const float* rp = base + (long)(gi - off) * nL;
    for (int c = tid * 32; c < nL; c += 256 * 32)
      __builtin_prefetch(rp + c, 0, 1);
  }
}

// ---------------------------------------------------------------------------
// Kernel 2: one wave32 per 16-anchor column tile.
//   D(16x16) = A(16x160, block-diagonal weights) x B(160x16, exp terms)
//   as 5 x v_wmma_f32_16x16x32_f16, f32 accumulate.
// Fragments are staged in LDS in the exact ISA register layout, so fragment
// loads are wide contiguous ds_load_b128s and the fill loops are branch-free.
// ---------------------------------------------------------------------------
__global__ void kde_wmma_kernel(const float* __restrict__ d0,
                                const float* __restrict__ d1,
                                const float* __restrict__ d2,
                                const float* __restrict__ migt,
                                const float* __restrict__ ws,
                                float* __restrict__ out) {
  __shared__ __attribute__((aligned(32))) _Float16 aL[5 * 16 * 32]; // [c][m][kk]
  __shared__ __attribute__((aligned(32))) _Float16 bL[5 * 16 * 32]; // [c][n][kk]
  __shared__ _Float16 s_w[NSLOT];
  __shared__ int      s_gi[NSLOT];
  __shared__ float    s_iv[NSLOT];

  const int lane = threadIdx.x;  // blockDim.x == 32 (one wave32)

  for (int i = lane; i < NSLOT; i += 32) {
    s_w[i]  = (_Float16)ws[WS_W + i];
    s_gi[i] = ((const int*)(ws + WS_GI))[i];
    s_iv[i] = ws[WS_IV + i];
  }
  __syncthreads();

  // Map tile -> pyramid level
  const int tile = blockIdx.x;
  const float* dptr; int nL, off, cb;
  if (tile < 256)      { dptr = d0; nL = 4096; off = 0;    cb = tile * 16; }
  else if (tile < 320) { dptr = d1; nL = 1024; off = 4096; cb = (tile - 256) * 16; }
  else                 { dptr = d2; nL = 256;  off = 5120; cb = (tile - 320) * 16; }

  // ---- Stage A (block-diagonal weights), branch-free: 80 entries/lane ----
  for (int t = lane; t < 5 * 16 * 32; t += 32) {
    int c = t >> 9, r = t & 511, m = r >> 5, kk = r & 31;
    int slot = (c << 5) + kk;
    _Float16 v = s_w[slot];
    aL[t] = ((slot / TOPK) == m) ? v : (_Float16)0.0f;
  }

  // ---- Stage B (exp terms), branch-free clamped-row gather: 80/lane ----
  for (int t = lane; t < 5 * 16 * 32; t += 32) {
    int c = t >> 9, r = t & 511, n = r >> 5, kk = r & 31;
    int slot = (c << 5) + kk;
    int gi = s_gi[slot];
    int inlvl = (gi >= off) & (gi < off + nL);       // slot's anchor in level?
    int rrow = inlvl ? (gi - off) : 0;               // row 0 is always valid
    float d = dptr[(long)rrow * nL + cb + n];        // unconditional load
    float e = __expf(-d * s_iv[slot]);
    bL[t] = inlvl ? (_Float16)e : (_Float16)0.0f;
  }
  __syncthreads();

  const int n    = lane & 15;   // output column
  const int hi   = lane >> 4;   // half-wave selector
  const int gcol = off + cb + n;

  v8f acc = {};
  #pragma unroll
  for (int c = 0; c < 5; ++c) {
    // A lane layout: elems 0..7 -> kk [8hi, 8hi+8), elems 8..15 -> kk [16+8hi, 16+8hi+8)
    const _Float16* ap = &aL[(c * 16 + n) * 32];
    v8h alo = *(const v8h*)(ap + 8 * hi);
    v8h ahi = *(const v8h*)(ap + 16 + 8 * hi);
    v16h A = __builtin_shufflevector(alo, ahi, 0, 1, 2, 3, 4, 5, 6, 7,
                                     8, 9, 10, 11, 12, 13, 14, 15);
    // B lane layout: same as A (elems 0..7 -> kk [8hi, 8hi+8), elems 8..15 -> kk [16+8hi, 16+8hi+8))
    const _Float16* bp = &bL[(c * 16 + n) * 32];
    v8h blo = *(const v8h*)(bp + 8 * hi);
    v8h bhi = *(const v8h*)(bp + 16 + 8 * hi);
    v16h B = __builtin_shufflevector(blo, bhi, 0, 1, 2, 3, 4, 5, 6, 7,
                                     8, 9, 10, 11, 12, 13, 14, 15);

    // (neg_a, A, neg_b, B, c_mod, C, reuse_a, reuse_b)
    acc = __builtin_amdgcn_wmma_f32_16x16x32_f16(false, A, false, B,
                                                 (short)0, acc, false, false);
  }

  // D layout: VGPR r -> row r + 8*hi, col = lane&15. Fuse mask_in_gts here.
  #pragma unroll
  for (int r = 0; r < 8; ++r) {
    int  orow = r + hi * 8;
    long o    = (long)orow * N_ANCH + gcol;
    out[o] = acc[r] * migt[o];
  }
}

// ---------------------------------------------------------------------------
// Kernel 3: in-place per-row max-normalization  p / (max(p) + eps).
// ---------------------------------------------------------------------------
__global__ void kde_norm_kernel(float* __restrict__ out) {
  const int row = blockIdx.x;
  const int tid = threadIdx.x;
  float* p = out + (long)row * N_ANCH;

  __shared__ float s_m[256];
  float m = -__builtin_inff();
  for (int i = tid; i < N_ANCH; i += 256) m = fmaxf(m, p[i]);
  s_m[tid] = m;
  __syncthreads();
  for (int s = 128; s > 0; s >>= 1) {
    if (tid < s) s_m[tid] = fmaxf(s_m[tid], s_m[tid + s]);
    __syncthreads();
  }
  const float scale = 1.0f / (s_m[0] + EPS_F);
  for (int i = tid; i < N_ANCH; i += 256) p[i] *= scale;  // each elem owned by one thread
}

// ---------------------------------------------------------------------------
extern "C" void kernel_launch(void* const* d_in, const int* in_sizes, int n_in,
                              void* d_out, int out_size, void* d_ws, size_t ws_size,
                              hipStream_t stream) {
  const float* am   = (const float*)d_in[0];  // (2,8,5376)
  const float* gtb  = (const float*)d_in[1];  // (2,8,4)
  const float* mgt  = (const float*)d_in[2];  // (2,8,1)
  const float* migt = (const float*)d_in[3];  // (2,8,5376)
  const float* dl0  = (const float*)d_in[4];  // (4096,4096)
  const float* dl1  = (const float*)d_in[5];  // (1024,1024)
  const float* dl2  = (const float*)d_in[6];  // (256,256)
  float*       out  = (float*)d_out;          // (2,8,5376)
  float*       ws   = (float*)d_ws;           // needs 1920 bytes

  kde_topk_kernel<<<NROW, 256, 0, stream>>>(am, gtb, mgt, dl0, dl1, dl2, ws);
  kde_wmma_kernel<<<336, 32, 0, stream>>>(dl0, dl1, dl2, migt, ws, out);
  kde_norm_kernel<<<NROW, 256, 0, stream>>>(out);
}
